// CFPLM_Model_80693845557547
// MI455X (gfx1250) — matrix operand order repaired and checked by reading, not compiled, not run on hardware
//
#include <hip/hip_runtime.h>
#include <hip/hip_bf16.h>
#include <math.h>
#include <stdint.h>

typedef __attribute__((ext_vector_type(16))) _Float16 v16h;
typedef __attribute__((ext_vector_type(8)))  float    v8f;
typedef __attribute__((ext_vector_type(4)))  int      v4i;

#define EMB   256
#define HEADS 4
#define HDIM  64
#define LN_EPS 1e-5f

// ---------------- CDNA5 async global->LDS copy (ASYNCcnt path) ----------------
#if defined(__has_builtin)
#  if __has_builtin(__builtin_amdgcn_global_load_async_to_lds_b128)
#    define HAS_ASYNC_B128 1
#  endif
#  if __has_builtin(__builtin_amdgcn_s_wait_asynccnt)
#    define HAS_WAIT_ASYNC 1
#  endif
#endif
#ifndef HAS_ASYNC_B128
#  define HAS_ASYNC_B128 0
#endif

typedef __attribute__((address_space(1))) v4i gas_v4i;   // global int4
typedef __attribute__((address_space(3))) v4i las_v4i;   // LDS int4

__device__ __forceinline__ void async_copy_b128(const void* g, void* l)
{
#if HAS_ASYNC_B128
    // generic LDS pointer: low 32 bits == LDS offset (aperture in high bits)
    __builtin_amdgcn_global_load_async_to_lds_b128(
        (gas_v4i*)(uintptr_t)g, (las_v4i*)(uintptr_t)l, 0, 0);
#else
    *(float4*)l = *(const float4*)g;
#endif
}

__device__ __forceinline__ void wait_async_lds()
{
#if HAS_ASYNC_B128
#  ifdef HAS_WAIT_ASYNC
    __builtin_amdgcn_s_wait_asynccnt(0);
#  else
    asm volatile("s_wait_asynccnt 0x0" ::: "memory");
#  endif
#endif
}

__device__ __forceinline__ float atomAddF(float* p, float v) {
    return __hip_atomic_fetch_add(p, v, __ATOMIC_RELAXED, __HIP_MEMORY_SCOPE_AGENT);
}

// ============================================================================
// WMMA GEMM:  C[M,N] = act( alpha * A[M,K] @ B + bias )
//   A: row-major (lda).  B: transB ? [N,K](ldb) : [K,N](ldb).  C: row-major ldc.
// Block = 2 waves (32x2). Block tile: 16(M) x 64(N); each wave does 16x32 with
// two v_wmma_f32_16x16x32_f16. A (16x32) and B (64x32, k-contiguous) tiles are
// staged in LDS; interior path uses GLOBAL_LOAD_ASYNC_TO_LDS_B128 (ASYNCcnt).
// ============================================================================
__global__ void gemm_wmma(const float* __restrict__ A, const float* __restrict__ B,
                          const float* __restrict__ bias, float* __restrict__ C,
                          int M, int N, int K, int lda, int ldb, int ldc,
                          int transB, float alpha, int relu)
{
    __shared__ float As[16][36];      // [m][k], padded: row = 144B (16B aligned)
    __shared__ float Bs[64][36];      // [n][k]
    const int lane   = threadIdx.x;          // 0..31
    const int wv     = threadIdx.y;          // 0..1
    const int tid    = wv * 32 + lane;       // 0..63
    const int m0     = blockIdx.y * 16;
    const int nblock = blockIdx.x * 64;
    const int lh     = lane >> 4;            // half-wave select
    const int l15    = lane & 15;

    v8f acc[2] = {{}, {}};

    for (int k0 = 0; k0 < K; k0 += 32) {
        const bool kFull = (k0 + 32 <= K);
        const bool mFull = (m0 + 16 <= M);
        const bool nFull = (nblock + 64 <= N);

        // ---------- stage A tile (16 x 32) ----------
        if (mFull && kFull) {
            #pragma unroll
            for (int it = 0; it < 2; ++it) {
                int f = tid + it * 64;
                int r = f >> 3, c4 = (f & 7) << 2;
                async_copy_b128(A + (size_t)(m0 + r) * lda + k0 + c4, &As[r][c4]);
            }
        } else {
            #pragma unroll
            for (int it = 0; it < 2; ++it) {
                int f = tid + it * 64;
                int r = f >> 3, c4 = (f & 7) << 2;
                int gm = m0 + r;
                #pragma unroll
                for (int j = 0; j < 4; ++j) {
                    int gk = k0 + c4 + j;
                    As[r][c4 + j] = (gm < M && gk < K) ? A[(size_t)gm * lda + gk] : 0.0f;
                }
            }
        }
        // ---------- stage B tile into Bs[n][k] (64 x 32) ----------
        if (transB) {
            if (nFull && kFull) {
                #pragma unroll
                for (int it = 0; it < 8; ++it) {
                    int g = tid + it * 64;
                    int n = g >> 3, c4 = (g & 7) << 2;
                    async_copy_b128(B + (size_t)(nblock + n) * ldb + k0 + c4, &Bs[n][c4]);
                }
            } else {
                #pragma unroll
                for (int it = 0; it < 8; ++it) {
                    int g = tid + it * 64;
                    int n = g >> 3, c4 = (g & 7) << 2;
                    int gn = nblock + n;
                    #pragma unroll
                    for (int j = 0; j < 4; ++j) {
                        int gk = k0 + c4 + j;
                        Bs[n][c4 + j] = (gn < N && gk < K) ? B[(size_t)gn * ldb + gk] : 0.0f;
                    }
                }
            }
        } else {
            if (nFull && kFull) {
                #pragma unroll
                for (int it = 0; it < 8; ++it) {
                    int g = tid + it * 64;
                    int kk = g & 31, n4 = (g >> 5) << 2;
                    float4 v = *(const float4*)(B + (size_t)(k0 + kk) * ldb + nblock + n4);
                    Bs[n4 + 0][kk] = v.x;
                    Bs[n4 + 1][kk] = v.y;
                    Bs[n4 + 2][kk] = v.z;
                    Bs[n4 + 3][kk] = v.w;
                }
            } else {
                #pragma unroll
                for (int it = 0; it < 8; ++it) {
                    int g = tid + it * 64;
                    int kk = g & 31, n4 = (g >> 5) << 2;
                    int gk = k0 + kk;
                    #pragma unroll
                    for (int j = 0; j < 4; ++j) {
                        int gn = nblock + n4 + j;
                        Bs[n4 + j][kk] = (gn < N && gk < K) ? B[(size_t)gk * ldb + gn] : 0.0f;
                    }
                }
            }
        }
        // prefetch next A tile (global_prefetch path)
        if (k0 + 32 < K) {
            __builtin_prefetch(A + (size_t)(m0 + (tid >> 3)) * lda + (k0 + 32) + ((tid & 7) << 2), 0, 3);
        }
        wait_async_lds();          // drain ASYNCcnt before cross-wave LDS visibility
        __syncthreads();

        // ---------- A fragment: e<8 -> k=lh*8+e ; e>=8 -> k=16+lh*8+(e-8) ----------
        v16h af;
        {
            float4 a0 = *(const float4*)&As[l15][lh * 8 + 0];
            float4 a1 = *(const float4*)&As[l15][lh * 8 + 4];
            float4 a2 = *(const float4*)&As[l15][16 + lh * 8 + 0];
            float4 a3 = *(const float4*)&As[l15][16 + lh * 8 + 4];
            af[0]=(_Float16)a0.x;  af[1]=(_Float16)a0.y;  af[2]=(_Float16)a0.z;  af[3]=(_Float16)a0.w;
            af[4]=(_Float16)a1.x;  af[5]=(_Float16)a1.y;  af[6]=(_Float16)a1.z;  af[7]=(_Float16)a1.w;
            af[8]=(_Float16)a2.x;  af[9]=(_Float16)a2.y;  af[10]=(_Float16)a2.z; af[11]=(_Float16)a2.w;
            af[12]=(_Float16)a3.x; af[13]=(_Float16)a3.y; af[14]=(_Float16)a3.z; af[15]=(_Float16)a3.w;
        }
        // ---------- B fragments: e<8 -> k=lh*8+e ; e>=8 -> k=16+lh*8+(e-8) ----------
        const int kb = lh * 8;
        #pragma unroll
        for (int t2 = 0; t2 < 2; ++t2) {
            int nloc = wv * 32 + t2 * 16 + l15;
            v16h bf;
            float4 b0 = *(const float4*)&Bs[nloc][kb + 0];
            float4 b1 = *(const float4*)&Bs[nloc][kb + 4];
            float4 b2 = *(const float4*)&Bs[nloc][16 + kb + 0];
            float4 b3 = *(const float4*)&Bs[nloc][16 + kb + 4];
            bf[0]=(_Float16)b0.x;  bf[1]=(_Float16)b0.y;  bf[2]=(_Float16)b0.z;  bf[3]=(_Float16)b0.w;
            bf[4]=(_Float16)b1.x;  bf[5]=(_Float16)b1.y;  bf[6]=(_Float16)b1.z;  bf[7]=(_Float16)b1.w;
            bf[8]=(_Float16)b2.x;  bf[9]=(_Float16)b2.y;  bf[10]=(_Float16)b2.z; bf[11]=(_Float16)b2.w;
            bf[12]=(_Float16)b3.x; bf[13]=(_Float16)b3.y; bf[14]=(_Float16)b3.z; bf[15]=(_Float16)b3.w;
            acc[t2] = __builtin_amdgcn_wmma_f32_16x16x32_f16(false, af, false, bf,
                                                             (short)0, acc[t2], false, false);
        }
        __syncthreads();
    }

    // C layout: VGPR r -> M = r + lh*8, N = lane&15
    #pragma unroll
    for (int t2 = 0; t2 < 2; ++t2) {
        int cn = nblock + wv * 32 + t2 * 16 + l15;
        if (cn < N) {
            #pragma unroll
            for (int r = 0; r < 8; ++r) {
                int cm = m0 + r + lh * 8;
                if (cm < M) {
                    float v = alpha * acc[t2][r];
                    if (bias) v += bias[cn];
                    if (relu) v = fmaxf(v, 0.0f);
                    C[(size_t)cm * ldc + cn] = v;
                }
            }
        }
    }
}

// ============================================================================
// Elementwise / reduction helpers
// ============================================================================
__global__ void fill_f32(float* __restrict__ p, long long n, float v)
{
    long long t = (long long)blockIdx.x * blockDim.x + threadIdx.x;
    if (t < n) p[t] = v;
}

__global__ void add_inplace(float* __restrict__ a, const float* __restrict__ b, long long n)
{
    long long t = (long long)blockIdx.x * blockDim.x + threadIdx.x;
    if (t < n) a[t] += b[t];
}

__global__ void deg_acc(float* __restrict__ deg, const int* __restrict__ col,
                        const float* __restrict__ w, int E)
{
    int e = blockIdx.x * blockDim.x + threadIdx.x;
    if (e < E) atomAddF(&deg[col[e]], w ? w[e] : 1.0f);
}

__global__ void deg_to_dinv(float* __restrict__ deg, int n)
{
    int i = blockIdx.x * blockDim.x + threadIdx.x;
    if (i < n) {
        float d = deg[i];
        deg[i] = (d > 0.0f) ? rsqrtf(fmaxf(d, 1e-12f)) : 0.0f;
    }
}

__global__ void edge_norm(float* __restrict__ nrm, const int* __restrict__ row,
                          const int* __restrict__ col, const float* __restrict__ w,
                          const float* __restrict__ dinv, int E)
{
    int e = blockIdx.x * blockDim.x + threadIdx.x;
    if (e < E) nrm[e] = dinv[row[e]] * (w ? w[e] : 1.0f) * dinv[col[e]];
}

// agg[col[e], :] += h[row[e], :] * nrm[e]   (thread = edge x 4-feature chunk)
__global__ void gcn_scatter(float* __restrict__ agg, const float* __restrict__ h,
                            const int* __restrict__ row, const int* __restrict__ col,
                            const float* __restrict__ nrm, int E)
{
    long long t = (long long)blockIdx.x * blockDim.x + threadIdx.x;
    if (t >= (long long)E * 64) return;
    int e = (int)(t >> 6);
    int c = (int)(t & 63) * 4;
    float wgt = nrm[e];
    const float4 hv = *(const float4*)(h + (size_t)row[e] * EMB + c);
    float* ap = agg + (size_t)col[e] * EMB + c;
    atomAddF(ap + 0, hv.x * wgt);
    atomAddF(ap + 1, hv.y * wgt);
    atomAddF(ap + 2, hv.z * wgt);
    atomAddF(ap + 3, hv.w * wgt);
}

// out = relu(agg + h * dinv^2 (self-loop) + b)
__global__ void gcn_finish(float* __restrict__ out, const float* __restrict__ agg,
                           const float* __restrict__ h, const float* __restrict__ dinv,
                           const float* __restrict__ b, int n)
{
    int t = blockIdx.x * blockDim.x + threadIdx.x;
    if (t >= n * EMB) return;
    int i = t >> 8, f = t & 255;
    float di = dinv[i];
    float v = agg[t] + h[t] * di * di + b[f];
    out[t] = fmaxf(v, 0.0f);
}

// in-place row softmax, one block per row
__global__ void softmax_rows(float* __restrict__ S, int N)
{
    __shared__ float red[256];
    float* rowp = S + (size_t)blockIdx.x * N;
    int t = threadIdx.x;
    float mx = -INFINITY;
    for (int j = t; j < N; j += 256) mx = fmaxf(mx, rowp[j]);
    red[t] = mx; __syncthreads();
    for (int s = 128; s > 0; s >>= 1) { if (t < s) red[t] = fmaxf(red[t], red[t + s]); __syncthreads(); }
    mx = red[0]; __syncthreads();
    float sum = 0.0f;
    for (int j = t; j < N; j += 256) { float e = expf(rowp[j] - mx); rowp[j] = e; sum += e; }
    red[t] = sum; __syncthreads();
    for (int s = 128; s > 0; s >>= 1) { if (t < s) red[t] += red[t + s]; __syncthreads(); }
    float inv = 1.0f / red[0];
    for (int j = t; j < N; j += 256) rowp[j] *= inv;
}

// out = LayerNorm(x + o) * g + b ; one block (256 == EMB threads) per row
__global__ void ln_residual(float* __restrict__ out, const float* __restrict__ x,
                            const float* __restrict__ o, const float* __restrict__ g,
                            const float* __restrict__ b)
{
    __shared__ float red[256];
    size_t base = (size_t)blockIdx.x * EMB;
    int t = threadIdx.x;
    float y = x[base + t] + o[base + t];
    red[t] = y; __syncthreads();
    for (int s = 128; s > 0; s >>= 1) { if (t < s) red[t] += red[t + s]; __syncthreads(); }
    float mu = red[0] * (1.0f / EMB); __syncthreads();
    float d = y - mu;
    red[t] = d * d; __syncthreads();
    for (int s = 128; s > 0; s >>= 1) { if (t < s) red[t] += red[t + s]; __syncthreads(); }
    float var = red[0] * (1.0f / EMB);
    out[base + t] = d * rsqrtf(var + LN_EPS) * g[t] + b[t];
}

// g[b, 0:256]=gather[idx[b,0]], g[b, 256:512]=gather[idx[b,1]]
__global__ void gather_pairs(float* __restrict__ g, const float* __restrict__ fr,
                             const float* __restrict__ fp, const int* __restrict__ idx,
                             int Nr, int B)
{
    long long t = (long long)blockIdx.x * blockDim.x + threadIdx.x;
    if (t >= (long long)B * 512) return;
    int b = (int)(t >> 9);
    int c = (int)(t & 511);
    int sel = idx[b * 2 + (c >> 8)];
    int f = c & 255;
    g[t] = (sel < Nr) ? fr[(size_t)sel * EMB + f] : fp[(size_t)(sel - Nr) * EMB + f];
}

// batch-norm stats over axis 0: one block per column
__global__ void bn_stats(float* __restrict__ stats, const float* __restrict__ h, int B)
{
    __shared__ float rs[256], rq[256];
    int j = blockIdx.x, t = threadIdx.x;
    float s = 0.0f, q = 0.0f;
    for (int r = t; r < B; r += 256) {
        float v = h[(size_t)r * EMB + j];
        s += v; q += v * v;
    }
    rs[t] = s; rq[t] = q; __syncthreads();
    for (int k = 128; k > 0; k >>= 1) {
        if (t < k) { rs[t] += rs[t + k]; rq[t] += rq[t + k]; }
        __syncthreads();
    }
    if (t == 0) {
        float mu = rs[0] / B;
        stats[j] = mu;
        stats[EMB + j] = rq[0] / B - mu * mu;
    }
}

__global__ void bn_apply_relu(float* __restrict__ h, const float* __restrict__ stats,
                              const float* __restrict__ g, const float* __restrict__ be,
                              long long n)
{
    long long t = (long long)blockIdx.x * blockDim.x + threadIdx.x;
    if (t >= n) return;
    int j = (int)(t & 255);
    float v = (h[t] - stats[j]) * rsqrtf(stats[EMB + j] + LN_EPS) * g[j] + be[j];
    h[t] = fmaxf(v, 0.0f);
}

// out[r] = sigmoid(dot(h[r,:], W2) + b2) ; one wave32 per row
__global__ void head_sigmoid(float* __restrict__ out, const float* __restrict__ h,
                             const float* __restrict__ W2, const float* __restrict__ b2,
                             int B)
{
    int lane = threadIdx.x & 31;
    int r = blockIdx.x * (blockDim.x >> 5) + (threadIdx.x >> 5);
    if (r >= B) return;
    float s = 0.0f;
    for (int j = lane; j < EMB; j += 32) s += h[(size_t)r * EMB + j] * W2[j];
    for (int o = 16; o > 0; o >>= 1) s += __shfl_down(s, o, 32);
    if (lane == 0) out[r] = 1.0f / (1.0f + expf(-(s + b2[0])));
}

__global__ void bce_loss(float* __restrict__ lossp, const float* __restrict__ out,
                         const float* __restrict__ lab, int B)
{
    __shared__ float red[256];
    int t = threadIdx.x;
    float s = 0.0f;
    for (int i = t; i < B; i += 256) {
        float o = fminf(fmaxf(out[i], 1e-7f), 1.0f - 1e-7f);
        float l = lab[i];
        s += l * logf(o) + (1.0f - l) * logf(1.0f - o);
    }
    red[t] = s; __syncthreads();
    for (int k = 128; k > 0; k >>= 1) { if (t < k) red[t] += red[t + k]; __syncthreads(); }
    if (t == 0) lossp[0] = -red[0] / B;
}

// ============================================================================
// Host orchestration
// ============================================================================
extern "C" void kernel_launch(void* const* d_in, const int* in_sizes, int n_in,
                              void* d_out, int out_size, void* d_ws, size_t ws_size,
                              hipStream_t stream)
{
    const int Fr = 640, Fp = 1024;
    const int Nr  = in_sizes[0] / Fr;
    const int Np  = in_sizes[1] / Fp;
    const int Er  = in_sizes[3];
    const int Ep  = in_sizes[5];
    const int Ei2 = in_sizes[6] / 2;     // total directed inter edges
    const int B   = in_sizes[8];
    const int Nt  = Nr + Np;
    const int Nmax = (Nr > Np) ? Nr : Np;

    auto F = [&](int i) { return (const float*)d_in[i]; };
    auto I = [&](int i) { return (const int*)d_in[i]; };

    // flattened param indices (setup_inputs dict order)
    const int P_RNA_AD = 9, P_RNA_INT = 13, P_PROT_INT = 17, P_PROT_AD = 21;
    const int P_RNA_CONV = 25, P_RP_CONV = 29, P_PROT_CONV = 33;
    const int P_CA1 = 37, P_CA2 = 47, P_MLP = 57;

    // workspace allocator
    char* wsb = (char*)d_ws;
    size_t off = 0;
    auto alloc = [&](size_t elems) -> float* {
        float* p = (float*)(wsb + off);
        off += ((elems * sizeof(float) + 255) / 256) * 256;
        return p;
    };
    int maxE = Er > Ep ? Er : Ep; if (Ei2 > maxE) maxE = Ei2;

    size_t g512_elems = (size_t)((B > Nr) ? B : Nr) * 512;
    float* g512     = alloc(g512_elems);            // adapter tmp / MLP concat
    float* rna_vec  = alloc((size_t)Nr * EMB);
    float* prot_vec = alloc((size_t)Np * EMB);
    float* rp_int   = alloc((size_t)Nt * EMB);
    float* hbuf     = alloc((size_t)Nt * EMB);      // GEMM output h
    float* aggbuf   = alloc((size_t)Nt * EMB);      // GCN aggregation / attn proj
    float* rna_gcn  = alloc((size_t)Nr * EMB);
    float* rp_gcn   = alloc((size_t)Nt * EMB);
    float* prot_gcn = alloc((size_t)Np * EMB);
    float* degb     = alloc((size_t)Nt);
    float* nrmb     = alloc((size_t)maxE);
    float* qb       = alloc((size_t)Nmax * EMB);
    float* kb       = alloc((size_t)Nmax * EMB);
    float* vb       = alloc((size_t)Nmax * EMB);
    float* ob       = alloc((size_t)Nmax * EMB);
    float* scores   = alloc((size_t)Nr * Np);
    float* attA_r   = alloc((size_t)Nr * EMB);
    float* attA_p   = alloc((size_t)Np * EMB);
    float* attB_r   = alloc((size_t)Nr * EMB);
    float* attB_p   = alloc((size_t)Np * EMB);
    float* hmlp     = alloc((size_t)B * EMB);
    float* stats    = alloc(512);
    (void)ws_size; (void)n_in;

    auto gemm = [&](const float* A, const float* Bm, const float* bias, float* Cm,
                    int M, int N, int K, int lda, int ldb, int ldc,
                    int transB, float alpha, int relu) {
        dim3 grid((N + 63) / 64, (M + 15) / 16);
        dim3 blk(32, 2);
        gemm_wmma<<<grid, blk, 0, stream>>>(A, Bm, bias, Cm, M, N, K,
                                            lda, ldb, ldc, transB, alpha, relu);
    };
    auto fill = [&](float* p, long long n, float v) {
        fill_f32<<<(int)((n + 255) / 256), 256, 0, stream>>>(p, n, v);
    };

    // ---------------- adapters: (x@W1+b1)@W2+b2 ----------------
    auto run_adapter = [&](const float* x, int n, int fin, int pb, float* dest) {
        gemm(x, F(pb + 0), F(pb + 1), g512, n, 512, fin, fin, 512, 512, 0, 1.0f, 0);
        gemm(g512, F(pb + 2), F(pb + 3), dest, n, EMB, 512, 512, EMB, EMB, 0, 1.0f, 0);
    };
    run_adapter(F(0), Nr, Fr, P_RNA_AD,  rna_vec);
    run_adapter(F(1), Np, Fp, P_PROT_AD, prot_vec);
    run_adapter(F(0), Nr, Fr, P_RNA_INT,  rp_int);
    run_adapter(F(1), Np, Fp, P_PROT_INT, rp_int + (size_t)Nr * EMB);

    // ---------------- 2-layer GCNs ----------------
    auto run_gcn = [&](const float* x, const int* row, const int* col, const float* val,
                       int E, int n, int pb, float* out) {
        fill(degb, n, 1.0f);                                      // self-loop weight
        deg_acc<<<(E + 255) / 256, 256, 0, stream>>>(degb, col, val, E);
        deg_to_dinv<<<(n + 255) / 256, 256, 0, stream>>>(degb, n);
        edge_norm<<<(E + 255) / 256, 256, 0, stream>>>(nrmb, row, col, val, degb, E);
        long long sc = (long long)E * 64;
        // layer 1
        gemm(x, F(pb + 0), nullptr, hbuf, n, EMB, EMB, EMB, EMB, EMB, 0, 1.0f, 0);
        fill(aggbuf, (long long)n * EMB, 0.0f);
        gcn_scatter<<<(int)((sc + 255) / 256), 256, 0, stream>>>(aggbuf, hbuf, row, col, nrmb, E);
        gcn_finish<<<(n * EMB + 255) / 256, 256, 0, stream>>>(aggbuf, aggbuf, hbuf, degb, F(pb + 1), n);
        // layer 2
        gemm(aggbuf, F(pb + 2), nullptr, hbuf, n, EMB, EMB, EMB, EMB, EMB, 0, 1.0f, 0);
        fill(aggbuf, (long long)n * EMB, 0.0f);
        gcn_scatter<<<(int)((sc + 255) / 256), 256, 0, stream>>>(aggbuf, hbuf, row, col, nrmb, E);
        gcn_finish<<<(n * EMB + 255) / 256, 256, 0, stream>>>(out, aggbuf, hbuf, degb, F(pb + 3), n);
    };
    run_gcn(rna_vec,  I(2), I(2) + Er,  F(3),    Er,  Nr, P_RNA_CONV,  rna_gcn);
    run_gcn(rp_int,   I(6), I(6) + Ei2, nullptr, Ei2, Nt, P_RP_CONV,   rp_gcn);
    run_gcn(prot_vec, I(4), I(4) + Ep,  F(5),    Ep,  Np, P_PROT_CONV, prot_gcn);

    const float* r_int_g = rp_gcn;
    const float* p_int_g = rp_gcn + (size_t)Nr * EMB;

    // ---------------- cross-attention ----------------
    auto run_ca = [&](const float* x1, int N1, const float* x2, int N2, int pb, float* dest) {
        gemm(x1, F(pb + 0), F(pb + 1), qb, N1, EMB, EMB, EMB, EMB, EMB, 0, 1.0f, 0); // Q
        gemm(x2, F(pb + 2), F(pb + 3), kb, N2, EMB, EMB, EMB, EMB, EMB, 0, 1.0f, 0); // K
        gemm(x2, F(pb + 4), F(pb + 5), vb, N2, EMB, EMB, EMB, EMB, EMB, 0, 1.0f, 0); // V
        for (int h = 0; h < HEADS; ++h) {
            // S = (q_h @ k_h^T) / sqrt(HDIM)
            gemm(qb + h * HDIM, kb + h * HDIM, nullptr, scores,
                 N1, N2, HDIM, EMB, EMB, N2, 1, 0.125f, 0);
            softmax_rows<<<N1, 256, 0, stream>>>(scores, N2);
            // o_h = S @ v_h
            gemm(scores, vb + h * HDIM, nullptr, ob + h * HDIM,
                 N1, HDIM, N2, N2, EMB, EMB, 0, 1.0f, 0);
        }
        gemm(ob, F(pb + 6), F(pb + 7), aggbuf, N1, EMB, EMB, EMB, EMB, EMB, 0, 1.0f, 0); // Wo
        ln_residual<<<N1, 256, 0, stream>>>(dest, x1, aggbuf, F(pb + 8), F(pb + 9));
    };
    run_ca(r_int_g,  Nr, p_int_g,  Np, P_CA1, attA_r);
    run_ca(p_int_g,  Np, r_int_g,  Nr, P_CA1, attA_p);
    run_ca(rna_gcn,  Nr, prot_gcn, Np, P_CA2, attB_r);
    run_ca(prot_gcn, Np, rna_gcn,  Nr, P_CA2, attB_p);

    add_inplace<<<(int)(((long long)Nr * EMB + 255) / 256), 256, 0, stream>>>(attA_r, attB_r, (long long)Nr * EMB);
    add_inplace<<<(int)(((long long)Np * EMB + 255) / 256), 256, 0, stream>>>(attA_p, attB_p, (long long)Np * EMB);

    // ---------------- pair MLP + loss ----------------
    long long gl = (long long)B * 512;
    gather_pairs<<<(int)((gl + 255) / 256), 256, 0, stream>>>(g512, attA_r, attA_p, I(7), Nr, B);
    gemm(g512, F(P_MLP + 0), F(P_MLP + 1), hmlp, B, EMB, 512, 512, EMB, EMB, 0, 1.0f, 0);
    bn_stats<<<EMB, 256, 0, stream>>>(stats, hmlp, B);
    long long hl = (long long)B * EMB;
    bn_apply_relu<<<(int)((hl + 255) / 256), 256, 0, stream>>>(hmlp, stats, F(P_MLP + 2), F(P_MLP + 3), hl);

    float* out = (float*)d_out;                       // [0..B): out, [B]: loss
    head_sigmoid<<<(B + 7) / 8, 256, 0, stream>>>(out, hmlp, F(P_MLP + 4), F(P_MLP + 5), B);
    bce_loss<<<1, 256, 0, stream>>>(out + B, out, F(8), B);
    (void)out_size;
}
